// LIIF_65455301591829
// MI455X (gfx1250) — compile-verified, matrix operand order checked by reading, not run on hardware
//
#include <hip/hip_runtime.h>
#include <hip/hip_bf16.h>

typedef __attribute__((ext_vector_type(16))) _Float16 v16h;
typedef __attribute__((ext_vector_type(8)))  _Float16 v8h;
typedef __attribute__((ext_vector_type(8)))  float    v8f;

#define C_IN 32
#define HW   256
#define K1PAD 320   // 292 padded to 10 k-tiles of 32
#define KHID 256

// ---------------------------------------------------------------------------
// Pack f32 weights -> f16, transposed to [N][Kpad] (B-fragment layout).
//   W1t: [256][320]  (rows k>=292 zero)
//   W2t,W3t,W4t: [256][256]
//   W5t: [16][256]   (cols n>=3 zero)
// ---------------------------------------------------------------------------
__global__ __launch_bounds__(256) void liif_pack_weights(
    const float* __restrict__ w1, const float* __restrict__ w2,
    const float* __restrict__ w3, const float* __restrict__ w4,
    const float* __restrict__ w5, _Float16* __restrict__ ws)
{
    const int idx = blockIdx.x * 256 + threadIdx.x;
    const int N1 = 256 * K1PAD;
    const int N2 = 256 * 256;
    if (idx < N1) {
        int n = idx / K1PAD, k = idx - n * K1PAD;
        ws[idx] = (k < 292) ? (_Float16)w1[k * 256 + n] : (_Float16)0.0f;
    } else if (idx < N1 + 3 * N2) {
        int r = idx - N1;
        int layer = r / N2;
        int e = r - layer * N2;
        int n = e >> 8, k = e & 255;
        const float* w = (layer == 0) ? w2 : (layer == 1) ? w3 : w4;
        ws[idx] = (_Float16)w[k * 256 + n];
    } else if (idx < N1 + 3 * N2 + 16 * 256) {
        int e = idx - N1 - 3 * N2;
        int n = e >> 8, k = e & 255;
        ws[idx] = (n < 3) ? (_Float16)w5[k * 3 + n] : (_Float16)0.0f;
    }
}

// ---------------------------------------------------------------------------
// Fragment load: works for A (rbase = m-tile row base, from LDS) and for
// B stored as Wt[n][k] (rbase = n-tile column base, from global/L2).
// 16-bit 16x32 layout: lane<16 holds K [0..8)+[16..24), lane>=16 holds
// K [8..16)+[24..32) of row/col (lane&15).  Two 16B chunks per lane.
// ---------------------------------------------------------------------------
__device__ __forceinline__ v16h load_tile(const _Float16* __restrict__ p,
                                          int stride, int rbase, int kbase,
                                          int lane)
{
    const int r    = rbase + (lane & 15);
    const int half = (lane >> 4) * 8;
    const _Float16* q = p + r * stride + kbase + half;
    union { v16h v; v8h h[2]; } u;
    u.h[0] = *(const v8h*)(q);        // K offsets [half, half+8)
    u.h[1] = *(const v8h*)(q + 16);   // K offsets [16+half, 16+half+8)
    return u.v;
}

// C/D 16x16 f32 layout: VGPR r -> row (r + 8*(lane>=16)), col (lane&15).
__device__ __forceinline__ void store_act(_Float16* __restrict__ outLds,
                                          v8f acc, const float* __restrict__ bias,
                                          int mbase, int nbase, int lane)
{
    const int col   = nbase + (lane & 15);
    const int rbase = mbase + ((lane >> 4) << 3);
    const float bn  = bias[col];
#pragma unroll
    for (int r = 0; r < 8; ++r) {
        float v = fmaxf(acc[r] + bn, 0.0f);        // bias + ReLU
        outLds[(rbase + r) * KHID + col] = (_Float16)v;
    }
}

// One dense layer: 32 rows (2 m-tiles) x 256 cols.  Each of the 8 waves owns
// 2 n-tiles; B loaded once per (nt,kt), reused across both m-tiles.
template <int KPAD, int KT>
__device__ __forceinline__ void mlp_layer(const _Float16* __restrict__ inLds,
                                          const _Float16* __restrict__ Wt,
                                          const float* __restrict__ bias,
                                          _Float16* __restrict__ outLds,
                                          int lane, int wv)
{
    const int nt0 = wv * 32;                        // two n-tiles: nt0, nt0+16
    __builtin_prefetch(Wt + (nt0 + lane) * KPAD, 0, 1);   // pull B panel via L2
    v8f acc00 = {}, acc01 = {}, acc10 = {}, acc11 = {};
#pragma unroll
    for (int kt = 0; kt < KT; ++kt) {
        const int kb = kt * 32;
        v16h a0 = load_tile(inLds, KPAD, 0,        kb, lane);
        v16h a1 = load_tile(inLds, KPAD, 16,       kb, lane);
        v16h b0 = load_tile(Wt,    KPAD, nt0,      kb, lane);
        v16h b1 = load_tile(Wt,    KPAD, nt0 + 16, kb, lane);
        acc00 = __builtin_amdgcn_wmma_f32_16x16x32_f16(false, a0, false, b0,
                                                       (short)0, acc00, false, false);
        acc01 = __builtin_amdgcn_wmma_f32_16x16x32_f16(false, a0, false, b1,
                                                       (short)0, acc01, false, false);
        acc10 = __builtin_amdgcn_wmma_f32_16x16x32_f16(false, a1, false, b0,
                                                       (short)0, acc10, false, false);
        acc11 = __builtin_amdgcn_wmma_f32_16x16x32_f16(false, a1, false, b1,
                                                       (short)0, acc11, false, false);
    }
    store_act(outLds, acc00, bias, 0,  nt0,      lane);
    store_act(outLds, acc01, bias, 0,  nt0 + 16, lane);
    store_act(outLds, acc10, bias, 16, nt0,      lane);
    store_act(outLds, acc11, bias, 16, nt0 + 16, lane);
}

// ---------------------------------------------------------------------------
// Fused LIIF kernel: 8 queries x 4 shifts = 32 rows per workgroup.
// Row index m = shift*8 + qlocal.
// ---------------------------------------------------------------------------
__global__ __launch_bounds__(256) void liif_fused(
    const float* __restrict__ features,   // [32][256][256]
    const float* __restrict__ locations,  // [256][256][2]
    const float* __restrict__ cells,      // [256][256][2]
    const _Float16* __restrict__ W1t, const float* __restrict__ b1,
    const _Float16* __restrict__ W2t, const float* __restrict__ b2,
    const _Float16* __restrict__ W3t, const float* __restrict__ b3,
    const _Float16* __restrict__ W4t, const float* __restrict__ b4,
    const _Float16* __restrict__ W5t, const float* __restrict__ b5,
    float* __restrict__ out)              // [256][256][3]
{
    __shared__ __align__(16) _Float16 Xb[32 * K1PAD];  // 20 KB
    __shared__ __align__(16) _Float16 A1[32 * KHID];   // 16 KB
    __shared__ __align__(16) _Float16 A2[32 * KHID];   // 16 KB
    __shared__ float s_relx[32], s_rely[32];
    __shared__ float s_pred[32][4];

    const int t    = threadIdx.x;
    const int lane = t & 31;
    const int wv   = t >> 5;

    // ------------------ prep: build X (32 x 320 f16) ------------------
    {
        const int row = t >> 3;      // 0..31
        const int idx = t & 7;       // 8 threads cooperate per row
        const int s   = row >> 3;    // shift 0..3
        const int q   = row & 7;
        const int qg  = blockIdx.x * 8 + q;
        const int qi  = qg >> 8, qj = qg & 255;
        const float loc0 = locations[(qi * 256 + qj) * 2 + 0];
        const float loc1 = locations[(qi * 256 + qj) * 2 + 1];
        const float c0   = cells[(qi * 256 + qj) * 2 + 0];
        const float c1   = cells[(qi * 256 + qj) * 2 + 1];
        const float sh0  = (s & 2) ? 1.0f : -1.0f;
        const float sh1  = (s & 1) ? 1.0f : -1.0f;
        const float rpix = 1.0f / 256.0f;
        float lx = fminf(fmaxf(loc0 + sh0 * rpix + 1e-6f, -1.0f + 1e-6f), 1.0f - 1e-6f);
        float ly = fminf(fmaxf(loc1 + sh1 * rpix + 1e-6f, -1.0f + 1e-6f), 1.0f - 1e-6f);
        int iy = (int)roundf(((lx + 1.0f) * 256.0f - 1.0f) * 0.5f);
        int ix = (int)roundf(((ly + 1.0f) * 256.0f - 1.0f) * 0.5f);
        iy = iy < 0 ? 0 : (iy > 255 ? 255 : iy);
        ix = ix < 0 ? 0 : (ix > 255 ? 255 : ix);
        int ys[3], xs[3];
#pragma unroll
        for (int d = 0; d < 3; ++d) {
            int yy = iy + d - 1;  ys[d] = (yy < 0) ? 1 : ((yy > 255) ? 254 : yy);
            int xx = ix + d - 1;  xs[d] = (xx < 0) ? 1 : ((xx > 255) ? 254 : xx);
        }
        for (int k = idx; k < 288; k += 8) {            // 3x3-unfold gather
            int c  = k / 9, p = k - c * 9;
            int di = p / 3, dj = p - di * 3;
            Xb[row * K1PAD + k] =
                (_Float16)features[(c << 16) + (ys[di] << 8) + xs[dj]];
        }
        if (idx == 0) {
            float qy = -1.0f + (2.0f * (float)iy + 1.0f) * rpix;
            float qx = -1.0f + (2.0f * (float)ix + 1.0f) * rpix;
            float rlx = (loc0 - qy) * 256.0f;
            float rly = (loc1 - qx) * 256.0f;
            s_relx[row] = rlx;
            s_rely[row] = rly;
            Xb[row * K1PAD + 288] = (_Float16)rlx;
            Xb[row * K1PAD + 289] = (_Float16)rly;
            Xb[row * K1PAD + 290] = (_Float16)(c0 * 256.0f);
            Xb[row * K1PAD + 291] = (_Float16)(c1 * 256.0f);
        }
        for (int k = 292 + idx; k < K1PAD; k += 8)      // zero K padding
            Xb[row * K1PAD + k] = (_Float16)0.0f;
    }
    __syncthreads();

    // ------------------ 4 hidden layers on WMMA ------------------
    mlp_layer<K1PAD, 10>(Xb, W1t, b1, A1, lane, wv);
    __syncthreads();
    mlp_layer<KHID, 8>(A1, W2t, b2, A2, lane, wv);
    __syncthreads();
    mlp_layer<KHID, 8>(A2, W3t, b3, A1, lane, wv);
    __syncthreads();
    mlp_layer<KHID, 8>(A1, W4t, b4, A2, lane, wv);
    __syncthreads();

    // ------------------ layer 5: 32 x 3 (padded to 16) ------------------
    if (wv < 2) {                       // wave-uniform branch, EXEC stays full
        v8f acc = {};
#pragma unroll
        for (int kt = 0; kt < 8; ++kt) {
            v16h a = load_tile(A2,  KHID, wv * 16, kt * 32, lane);
            v16h b = load_tile(W5t, KHID, 0,       kt * 32, lane);
            acc = __builtin_amdgcn_wmma_f32_16x16x32_f16(false, a, false, b,
                                                         (short)0, acc, false, false);
        }
        const int col   = lane & 15;
        const int rbase = wv * 16 + ((lane >> 4) << 3);
        if (col < 3) {
            const float bn = b5[col];
#pragma unroll
            for (int r = 0; r < 8; ++r)
                s_pred[rbase + r][col] = acc[r] + bn;
        }
    }
    __syncthreads();

    // ------------------ area-weighted blend of 4 shifts ------------------
    if (t < 24) {
        const int q  = t / 3;
        const int ch = t - q * 3;
        float a0 = fabsf(s_relx[0 * 8 + q] * s_rely[0 * 8 + q]) + 1e-9f;
        float a1 = fabsf(s_relx[1 * 8 + q] * s_rely[1 * 8 + q]) + 1e-9f;
        float a2 = fabsf(s_relx[2 * 8 + q] * s_rely[2 * 8 + q]) + 1e-9f;
        float a3 = fabsf(s_relx[3 * 8 + q] * s_rely[3 * 8 + q]) + 1e-9f;
        float tot = a0 + a1 + a2 + a3;
        float v = s_pred[0 * 8 + q][ch] * a3 + s_pred[1 * 8 + q][ch] * a2 +
                  s_pred[2 * 8 + q][ch] * a1 + s_pred[3 * 8 + q][ch] * a0;
        out[(blockIdx.x * 8 + q) * 3 + ch] = v / tot;
    }
}

// ---------------------------------------------------------------------------
extern "C" void kernel_launch(void* const* d_in, const int* in_sizes, int n_in,
                              void* d_out, int out_size, void* d_ws, size_t ws_size,
                              hipStream_t stream)
{
    const float* features  = (const float*)d_in[0];
    const float* locations = (const float*)d_in[1];
    const float* cells     = (const float*)d_in[2];
    const float* w1 = (const float*)d_in[3];  const float* b1 = (const float*)d_in[4];
    const float* w2 = (const float*)d_in[5];  const float* b2 = (const float*)d_in[6];
    const float* w3 = (const float*)d_in[7];  const float* b3 = (const float*)d_in[8];
    const float* w4 = (const float*)d_in[9];  const float* b4 = (const float*)d_in[10];
    const float* w5 = (const float*)d_in[11]; const float* b5 = (const float*)d_in[12];
    float* out = (float*)d_out;

    _Float16* ws  = (_Float16*)d_ws;
    _Float16* W1t = ws;                         // 256*320
    _Float16* W2t = W1t + 256 * K1PAD;          // 256*256
    _Float16* W3t = W2t + 256 * 256;
    _Float16* W4t = W3t + 256 * 256;
    _Float16* W5t = W4t + 256 * 256;            // 16*256

    const int total = 256 * K1PAD + 3 * 256 * 256 + 16 * 256;
    liif_pack_weights<<<(total + 255) / 256, 256, 0, stream>>>(w1, w2, w3, w4, w5, ws);

    const int nblocks = (256 * 256) / 8;        // 8 queries per workgroup
    liif_fused<<<nblocks, 256, 0, stream>>>(features, locations, cells,
                                            W1t, b1, W2t, b2, W3t, b3,
                                            W4t, b4, W5t, b5, out);
}
